// Block_35837207118566
// MI455X (gfx1250) — compile-verified, matrix operand order checked by reading, not run on hardware
//
#include <hip/hip_runtime.h>
#include <hip/hip_bf16.h>

// ---------------------------------------------------------------------------
// CDNA5 (gfx1250) interval-bound causal attention, bf16 WMMA (16x16x32, f32 acc)
// ---------------------------------------------------------------------------

typedef __attribute__((ext_vector_type(16))) __bf16 bf16x16;
typedef __attribute__((ext_vector_type(8)))  __bf16 bf16x8;
typedef __attribute__((ext_vector_type(8)))  float  f32x8;

#define B_   4
#define T_   1024
#define C_   768
#define H_   12
#define D_   64
#define MROWS (B_ * T_)        // 4096
#define NQKV  (3 * C_)         // 2304
static const size_t HS = (size_t)B_ * H_ * T_ * D_;   // 3,145,728 elems per head-slot

__device__ __forceinline__ f32x8 wmma_bf16(bf16x16 a, bf16x16 b, f32x8 c) {
  // D = A(16x32 bf16) x B(32x16 bf16) + C(16x16 f32)
  return __builtin_amdgcn_wmma_f32_16x16x32_bf16(false, a, false, b, (short)0, c,
                                                 false, false);
}

union Frag16 { bf16x16 v; bf16x8 h[2]; };

// Row-major fragment load. ISA 7.12.2: 16-bit A 16x32 — lane L holds row M=L%16;
// elements e<8 at K = lanehi*8 + e, e>=8 at K = 16 + lanehi*8 + (e-8).
// Two contiguous 16B (8 x bf16) loads per lane. Same pattern assumed for B
// (lane = column N, elements over K) so "rowBase" = tile's first row/col.
__device__ __forceinline__ bf16x16 load_frag(const __bf16* base, int ld,
                                             int rowBase, int k0, int lane) {
  const int r = rowBase + (lane & 15);
  const __bf16* p = base + (size_t)r * ld + k0 + ((lane >> 4) << 3);
  Frag16 f;
  f.h[0] = *(const bf16x8*)(p);
  f.h[1] = *(const bf16x8*)(p + 16);
  return f.v;
}

// A-fragment read through merged-head addressing: logical row i=(b,t), logical
// col c=h*64+d over tensor stored [B,H,T,D]. 32-aligned K chunks never cross a
// head boundary, so each 8-half stays contiguous.
__device__ __forceinline__ bf16x16 load_afrag_heads(const __bf16* A, int rowBase,
                                                    int k0, int lane) {
  const int i = rowBase + (lane & 15);
  const int b = i >> 10, t = i & 1023;
  const int c0 = k0 + ((lane >> 4) << 3);
  Frag16 f;
#pragma unroll
  for (int half = 0; half < 2; ++half) {
    const int c = c0 + half * 16;
    const int h = c >> 6, d = c & 63;
    const __bf16* p = A + (((size_t)(b * H_ + h) * T_ + t) * D_ + d);
    f.h[half] = *(const bf16x8*)p;
  }
  return f.v;
}

// ---------------------------------------------------------------------------
// Kernel 1: f32 -> bf16 conversions (x, midpoint, radius, W, |W|, Wproj)
// ---------------------------------------------------------------------------
__global__ void __launch_bounds__(256)
convert_kernel(const float* X, const float* XL, const float* XU,
               const float* W, const float* WP,
               __bf16* XB, __bf16* MB, __bf16* RB,
               __bf16* WB, __bf16* WAB, __bf16* WPB) {
  const int idx = blockIdx.x * 256 + threadIdx.x;
  const int NX  = MROWS * C_;   // 3,145,728
  const int NW  = NQKV * C_;    // 1,769,472
  const int NWP = C_ * C_;      //   589,824
  if (idx < NX) {
    float xl = XL[idx], xu = XU[idx];
    XB[idx] = (__bf16)X[idx];
    MB[idx] = (__bf16)(0.5f * (xl + xu));   // midpoint
    RB[idx] = (__bf16)(0.5f * (xu - xl));   // radius (>= 0)
  }
  if (idx < NW) {
    float w = W[idx];
    WB[idx]  = (__bf16)w;
    WAB[idx] = (__bf16)fabsf(w);
  }
  if (idx < NWP) WPB[idx] = (__bf16)WP[idx];
}

// ---------------------------------------------------------------------------
// Kernel 2: QKV projection.  S_x = x@W^T, S_m = m@W^T, S_r = r@|W|^T
//           exact = S_x ; lo = S_m - S_r ; hi = S_m + S_r
// Wave tile 16x32, block = 8 waves (4M x 2N) -> 64x64. grid (64, 36).
// Epilogue scatters into per-head slots; v-type outputs stored transposed
// [B,H,D,T] so attention's A*V B-operand loads contiguously over keys.
// ---------------------------------------------------------------------------
__global__ void __launch_bounds__(256)
qkv_gemm_kernel(const __bf16* X, const __bf16* Mm, const __bf16* R,
                const __bf16* W, const __bf16* Wa, __bf16* QKV9) {
  const int lane = threadIdx.x & 31;
  const int wave = threadIdx.x >> 5;
  const int row0 = blockIdx.x * 64 + (wave & 3) * 16;
  const int col0 = blockIdx.y * 64 + (wave >> 2) * 32;

  f32x8 ax[2], am[2], ar[2];
#pragma unroll
  for (int t = 0; t < 2; ++t)
#pragma unroll
    for (int e = 0; e < 8; ++e) { ax[t][e] = 0.f; am[t][e] = 0.f; ar[t][e] = 0.f; }

  for (int k0 = 0; k0 < C_; k0 += 32) {
    bf16x16 fx = load_frag(X,  C_, row0, k0, lane);
    bf16x16 fm = load_frag(Mm, C_, row0, k0, lane);
    bf16x16 fr = load_frag(R,  C_, row0, k0, lane);
#pragma unroll
    for (int t = 0; t < 2; ++t) {
      bf16x16 bw  = load_frag(W,  C_, col0 + t * 16, k0, lane);
      bf16x16 bwa = load_frag(Wa, C_, col0 + t * 16, k0, lane);
      ax[t] = wmma_bf16(fx, bw,  ax[t]);
      am[t] = wmma_bf16(fm, bw,  am[t]);
      ar[t] = wmma_bf16(fr, bwa, ar[t]);
    }
  }

  const int n = lane & 15, mh = (lane >> 4) << 3;
#pragma unroll
  for (int t = 0; t < 2; ++t) {
#pragma unroll
    for (int e = 0; e < 8; ++e) {
      const int i = row0 + mh + e;
      const int j = col0 + t * 16 + n;
      const int b = i >> 10, tt = i & 1023;
      const int which = j / C_;          // 0=q 1=k 2=v
      const int c = j - which * C_;
      const int h = c >> 6, d = c & 63;
      const float ex = ax[t][e];
      const float lo = am[t][e] - ar[t][e];
      const float hi = am[t][e] + ar[t][e];
      size_t off;
      if (which == 2) off = ((size_t)((b * H_ + h) * D_ + d)) * T_ + tt;  // V^T
      else            off = ((size_t)((b * H_ + h) * T_ + tt)) * D_ + d;
      QKV9[(size_t)which       * HS + off] = (__bf16)ex;
      QKV9[(size_t)(3 + which) * HS + off] = (__bf16)lo;
      QKV9[(size_t)(6 + which) * HS + off] = (__bf16)hi;
    }
  }
}

// ---------------------------------------------------------------------------
// Kernel 3: flash attention, grid (T/128, B*H, 5 variants).
// variant 0: exact(q,k)->@v ; 1: (ql,kl) ; 2: (ql,ku) ; 3: (qu,kl) ; 4: (qu,ku)
// Bound variants emit A@vl and A@vu. Per wave: 16 query rows, 32-key tiles.
// Online softmax done with __shfl_xor over 16-lane halves; P bounces through
// per-wave LDS to convert C-layout -> A-layout. No block-wide sync (waves have
// different causal trip counts).
// ---------------------------------------------------------------------------
__global__ void __launch_bounds__(256)
flash_attn_kernel(const __bf16* QKV9, __bf16* O9) {
  const int lane = threadIdx.x & 31;
  const int wave = threadIdx.x >> 5;
  const int variant = blockIdx.z;
  const int bh = blockIdx.y;
  const int qbase = blockIdx.x * 128 + wave * 16;

  const int qs = (variant == 0) ? 0 : ((variant <= 2) ? 3 : 6);
  const int ks = (variant == 0) ? 1 : (((variant & 1) != 0) ? 4 : 7);

  const __bf16* Q  = QKV9 + (size_t)qs * HS + (size_t)bh * T_ * D_;
  const __bf16* Kp = QKV9 + (size_t)ks * HS + (size_t)bh * T_ * D_;
  const __bf16* V0 = QKV9 + (size_t)(variant == 0 ? 2 : 5) * HS + (size_t)bh * D_ * T_;
  const __bf16* V1 = QKV9 + (size_t)(variant == 0 ? 2 : 8) * HS + (size_t)bh * D_ * T_;
  const int nOut = (variant == 0) ? 1 : 2;

  __shared__ __align__(16) __bf16 Plds[8][16][32];   // per-wave P tile, 8 KB

  const bf16x16 aq0 = load_frag(Q, D_, qbase, 0,  lane);
  const bf16x16 aq1 = load_frag(Q, D_, qbase, 32, lane);

  f32x8 acc[2][4];
  float mrow[8], lrow[8], scale[8];
#pragma unroll
  for (int o = 0; o < 2; ++o)
#pragma unroll
    for (int dt = 0; dt < 4; ++dt)
#pragma unroll
      for (int e = 0; e < 8; ++e) acc[o][dt][e] = 0.f;
#pragma unroll
  for (int e = 0; e < 8; ++e) { mrow[e] = -1e30f; lrow[e] = 0.f; }

  const int n = lane & 15, mh = (lane >> 4) << 3;
  const int ktiles = (qbase + 16 + 31) >> 5;

  for (int kt = 0; kt < ktiles; ++kt) {
    const int kb = kt << 5;

    // ---- scores S = (Q K^T) * 1/sqrt(64), causal-masked ----
    f32x8 S[2];
#pragma unroll
    for (int nt = 0; nt < 2; ++nt) {
      bf16x16 bk0 = load_frag(Kp, D_, kb + nt * 16, 0,  lane);
      bf16x16 bk1 = load_frag(Kp, D_, kb + nt * 16, 32, lane);
      f32x8 s;
#pragma unroll
      for (int e = 0; e < 8; ++e) s[e] = 0.f;
      s = wmma_bf16(aq0, bk0, s);
      s = wmma_bf16(aq1, bk1, s);
      S[nt] = s;
    }

    // ---- per-row online softmax (rows live in fixed VGPR slices e) ----
#pragma unroll
    for (int e = 0; e < 8; ++e) {
      const int qi = qbase + mh + e;
      float s0 = S[0][e] * 0.125f; if (kb + n      > qi) s0 = -1e30f;
      float s1 = S[1][e] * 0.125f; if (kb + 16 + n > qi) s1 = -1e30f;
      float rm = fmaxf(s0, s1);
      rm = fmaxf(rm, __shfl_xor(rm, 1, 32));
      rm = fmaxf(rm, __shfl_xor(rm, 2, 32));
      rm = fmaxf(rm, __shfl_xor(rm, 4, 32));
      rm = fmaxf(rm, __shfl_xor(rm, 8, 32));
      const float mn = fmaxf(mrow[e], rm);
      const float p0 = __expf(s0 - mn);
      const float p1 = __expf(s1 - mn);
      float ps = p0 + p1;
      ps += __shfl_xor(ps, 1, 32);
      ps += __shfl_xor(ps, 2, 32);
      ps += __shfl_xor(ps, 4, 32);
      ps += __shfl_xor(ps, 8, 32);
      const float sc = __expf(mrow[e] - mn);
      lrow[e] = lrow[e] * sc + ps;
      mrow[e] = mn;
      scale[e] = sc;
      Plds[wave][mh + e][n]      = (__bf16)p0;
      Plds[wave][mh + e][16 + n] = (__bf16)p1;
    }

    // ---- rescale accumulators ----
#pragma unroll
    for (int o = 0; o < 2; ++o) {
      if (o >= nOut) break;
#pragma unroll
      for (int dt = 0; dt < 4; ++dt)
#pragma unroll
        for (int e = 0; e < 8; ++e)
          acc[o][dt][e] = acc[o][dt][e] * scale[e];
    }

    // ---- A @ V : P (16x32) from LDS, V^T rows contiguous over keys ----
    const bf16x16 pa =
        load_frag((const __bf16*)&Plds[wave][0][0], 32, 0, 0, lane);
#pragma unroll
    for (int dt = 0; dt < 4; ++dt) {
      bf16x16 bv0 = load_frag(V0, T_, dt * 16, kb, lane);
      acc[0][dt] = wmma_bf16(pa, bv0, acc[0][dt]);
      if (nOut == 2) {
        bf16x16 bv1 = load_frag(V1, T_, dt * 16, kb, lane);
        acc[1][dt] = wmma_bf16(pa, bv1, acc[1][dt]);
      }
    }
  }

  // ---- normalize + store [B,H,T,D] bf16 ----
#pragma unroll
  for (int o = 0; o < 2; ++o) {
    if (o >= nOut) break;
    const int slot = (variant == 0) ? 0 : (2 * variant - 1 + o);
    __bf16* Od = O9 + (size_t)slot * HS + (size_t)bh * T_ * D_;
#pragma unroll
    for (int dt = 0; dt < 4; ++dt)
#pragma unroll
      for (int e = 0; e < 8; ++e) {
        const int t = qbase + mh + e;
        const int d = dt * 16 + n;
        Od[(size_t)t * D_ + d] = (__bf16)(acc[o][dt][e] / lrow[e]);
      }
  }
}

// ---------------------------------------------------------------------------
// Kernel 4: y_lower = min over 8 bound outputs, y_upper = max
// ---------------------------------------------------------------------------
__global__ void __launch_bounds__(256)
combine_kernel(const __bf16* O9, __bf16* Yl, __bf16* Yu) {
  const size_t idx = (size_t)blockIdx.x * 256 + threadIdx.x;
  if (idx >= HS) return;
  float mn = 1e30f, mx = -1e30f;
#pragma unroll
  for (int s = 1; s < 9; ++s) {
    const float v = (float)O9[(size_t)s * HS + idx];
    mn = fminf(mn, v);
    mx = fmaxf(mx, v);
  }
  Yl[idx] = (__bf16)mn;
  Yu[idx] = (__bf16)mx;
}

// ---------------------------------------------------------------------------
// Kernel 5: output projection (y, y_lower, y_upper) @ Wproj^T -> f32 d_out.
// grid (64, 12, 3); wave tile 16x32; A read with merged-head addressing.
// ---------------------------------------------------------------------------
__global__ void __launch_bounds__(256)
proj_gemm_kernel(const __bf16* Ye, const __bf16* Yl, const __bf16* Yu,
                 const __bf16* Wp, float* out) {
  const int lane = threadIdx.x & 31;
  const int wave = threadIdx.x >> 5;
  const int z = blockIdx.z;
  const __bf16* A = (z == 0) ? Ye : (z == 1) ? Yl : Yu;
  const int row0 = blockIdx.x * 64 + (wave & 3) * 16;
  const int col0 = blockIdx.y * 64 + (wave >> 2) * 32;

  f32x8 accv[2];
#pragma unroll
  for (int t = 0; t < 2; ++t)
#pragma unroll
    for (int e = 0; e < 8; ++e) accv[t][e] = 0.f;

  for (int k0 = 0; k0 < C_; k0 += 32) {
    bf16x16 fa = load_afrag_heads(A, row0, k0, lane);
#pragma unroll
    for (int t = 0; t < 2; ++t) {
      bf16x16 bw = load_frag(Wp, C_, col0 + t * 16, k0, lane);
      accv[t] = wmma_bf16(fa, bw, accv[t]);
    }
  }

  const int n = lane & 15, mh = (lane >> 4) << 3;
  float* dst = out + (size_t)z * MROWS * C_;
#pragma unroll
  for (int t = 0; t < 2; ++t)
#pragma unroll
    for (int e = 0; e < 8; ++e) {
      const int i = row0 + mh + e;
      const int j = col0 + t * 16 + n;
      dst[(size_t)i * C_ + j] = accv[t][e];
    }
}

// ---------------------------------------------------------------------------
// Host-side launcher
// ---------------------------------------------------------------------------
extern "C" void kernel_launch(void* const* d_in, const int* in_sizes, int n_in,
                              void* d_out, int out_size, void* d_ws, size_t ws_size,
                              hipStream_t stream) {
  const float* x  = (const float*)d_in[0];
  const float* xl = (const float*)d_in[1];
  const float* xu = (const float*)d_in[2];
  const float* w  = (const float*)d_in[3];
  const float* wp = (const float*)d_in[4];

  char* ws = (char*)d_ws;
  size_t off = 0;
  auto alloc = [&](size_t bytes) -> char* {
    char* p = ws + off;
    off += (bytes + 255) & ~(size_t)255;
    return p;
  };

  const size_t NX  = (size_t)MROWS * C_;
  const size_t NW  = (size_t)NQKV * C_;
  const size_t NWP = (size_t)C_ * C_;

  __bf16* XB   = (__bf16*)alloc(NX * 2);
  __bf16* MB   = (__bf16*)alloc(NX * 2);
  __bf16* RB   = (__bf16*)alloc(NX * 2);
  __bf16* WB   = (__bf16*)alloc(NW * 2);
  __bf16* WAB  = (__bf16*)alloc(NW * 2);
  __bf16* WPB  = (__bf16*)alloc(NWP * 2);
  __bf16* QKV9 = (__bf16*)alloc(9 * HS * 2);
  __bf16* O9   = (__bf16*)alloc(9 * HS * 2);
  __bf16* YL   = (__bf16*)alloc(HS * 2);
  __bf16* YU   = (__bf16*)alloc(HS * 2);
  (void)ws_size; (void)in_sizes; (void)n_in; (void)out_size;

  convert_kernel<<<(int)((NX + 255) / 256), 256, 0, stream>>>(
      x, xl, xu, w, wp, XB, MB, RB, WB, WAB, WPB);

  qkv_gemm_kernel<<<dim3(MROWS / 64, NQKV / 64), 256, 0, stream>>>(
      XB, MB, RB, WB, WAB, QKV9);

  flash_attn_kernel<<<dim3(T_ / 128, B_ * H_, 5), 256, 0, stream>>>(QKV9, O9);

  combine_kernel<<<(int)((HS + 255) / 256), 256, 0, stream>>>(O9, YL, YU);

  proj_gemm_kernel<<<dim3(MROWS / 64, C_ / 64, 3), 256, 0, stream>>>(
      O9 /* slot 0 = exact y */, YL, YU, WPB, (float*)d_out);
}
